// RegionProposalNetwork_49314814492805
// MI455X (gfx1250) — matrix operand order repaired and checked by reading, not compiled, not run on hardware
//
#include <hip/hip_runtime.h>

// ---------------- problem constants ----------------
#define BB      8
#define HH      38
#define WW      50
#define CIN     1024
#define HID     2048
#define NA      9
#define NPOS    (BB*HH*WW)          // 15200
#define NANCH   (HH*WW*NA)          // 17100
#define KTOT    (9*CIN)             // 9216
#define TOPK    2000
#define MAXOUT  300
#define IOU_T   0.7f
#define STRIDE_F 16.0f

#define MTILES  (NPOS/16)           // 950
#define KSTAGE  64                  // K values staged per LDS buffer
#define BSTRIDE 72                  // LDS column stride in bf16 (144B: 16B-aligned, bank-safe)

// ---------------- CDNA5 async-to-LDS availability ----------------
#if defined(__has_builtin)
# if __has_builtin(__builtin_amdgcn_global_load_async_to_lds_b128)
#  define HAVE_ASYNC_LDS 1
# endif
#endif
#ifndef HAVE_ASYNC_LDS
# define HAVE_ASYNC_LDS 0
#endif

#if HAVE_ASYNC_LDS
# if defined(__has_builtin) && __has_builtin(__builtin_amdgcn_s_wait_asynccnt)
#  define WAIT_ASYNC0() __builtin_amdgcn_s_wait_asynccnt(0)
# else
#  define WAIT_ASYNC0() asm volatile("s_wait_asynccnt 0x0" ::: "memory")
# endif
#endif

// lightweight workgroup barrier: DS-only fence (no LOADcnt drain) + split barrier
static __device__ inline void wg_barrier_lds() {
    asm volatile("s_wait_dscnt 0x0" ::: "memory");
#if defined(__has_builtin) && __has_builtin(__builtin_amdgcn_s_barrier_signal) && __has_builtin(__builtin_amdgcn_s_barrier_wait)
    __builtin_amdgcn_s_barrier_signal(-1);
    __builtin_amdgcn_s_barrier_wait(-1);
#else
    asm volatile("s_barrier_signal -1\n\ts_barrier_wait -1" ::: "memory");
#endif
}

// ---------------- vector types ----------------
typedef __attribute__((ext_vector_type(8)))  __bf16 v8bf;
typedef __attribute__((ext_vector_type(16))) __bf16 v16bf;
typedef __attribute__((ext_vector_type(8)))  float  v8f;
typedef int v4i32 __attribute__((vector_size(16)));

union AFrag { uint4 q[2]; v16bf v; };

// ---------------- helpers ----------------
static __device__ inline unsigned short f2bf(float f) {
    unsigned u = __float_as_uint(f);
    u += 0x7FFFu + ((u >> 16) & 1u);          // round to nearest even
    return (unsigned short)(u >> 16);
}
static __device__ inline v16bf lds_load16(const unsigned short* p) {
    v8bf lo = *reinterpret_cast<const v8bf*>(p);
    v8bf hi = *reinterpret_cast<const v8bf*>(p + 8);
    v16bf r;
#pragma unroll
    for (int i = 0; i < 8; ++i) { r[i] = lo[i]; r[i + 8] = hi[i]; }
    return r;
}
static __device__ inline uint4 and4(uint4 a, unsigned m) {
    a.x &= m; a.y &= m; a.z &= m; a.w &= m; return a;
}
static __device__ inline v16bf mask_frag(uint4 lo, uint4 hi, unsigned m) {
    AFrag ua;
    ua.q[0] = and4(lo, m);
    ua.q[1] = and4(hi, m);
    return ua.v;
}

#if HAVE_ASYNC_LDS
static __device__ inline void async_ld_b128(const unsigned short* g, unsigned short* l) {
    __builtin_amdgcn_global_load_async_to_lds_b128(
        (__attribute__((address_space(1))) v4i32*)(unsigned long long)g,
        (__attribute__((address_space(3))) v4i32*)(unsigned)(unsigned long long)l,
        0, 0);
}
#endif

// ---------------- kernel 1: fmap fp32 -> bf16 ----------------
__global__ __launch_bounds__(256)
void cvt_fmap_kernel(const float* __restrict__ in, unsigned short* __restrict__ out, int n) {
    for (int i = blockIdx.x * 256 + threadIdx.x; i < n; i += gridDim.x * 256)
        out[i] = f2bf(in[i]);
}

// ---------------- kernel 2: w1 [9*CIN][HID] fp32 -> packed [HID][KTOT] bf16 ----------------
__global__ __launch_bounds__(256)
void pack_w1_kernel(const float* __restrict__ w1, unsigned short* __restrict__ wp) {
    const int n = KTOT * HID;
    for (int i = blockIdx.x * 256 + threadIdx.x; i < n; i += gridDim.x * 256) {
        int k = i / HID;          // k = tap*CIN + cin
        int c = i % HID;          // output channel (coalesced read)
        wp[(size_t)c * KTOT + k] = f2bf(w1[i]);
    }
}

// ---------------- kernel 3: implicit-GEMM 3x3 conv via WMMA bf16 ----------------
// grid (ceil(950/8), 32), block 256 (8 waves). Wave -> 16 rows x 64 cols.
// B stage (64 cols x 64 K) in LDS, double buffered via async-to-LDS; 8 WMMAs/barrier.
__global__ __launch_bounds__(256)
void conv1_wmma_kernel(const unsigned short* __restrict__ fmap,  // bf16 [NPOS][CIN]
                       const unsigned short* __restrict__ wp,    // bf16 [HID][KTOT]
                       const float* __restrict__ b1,
                       unsigned short* __restrict__ hidden)      // bf16 [NPOS][HID]
{
    __shared__ unsigned short ldsB[2][64 * BSTRIDE];             // 2 x 9216 B

    const int lane = threadIdx.x & 31;
    const int wave = threadIdx.x >> 5;
    int mt = blockIdx.x * 8 + wave;
    if (mt >= MTILES) mt = MTILES - 1;          // clamp: duplicate tile, identical stores
    const int m0 = mt * 16;
    const int n0 = blockIdx.y * 64;
    const int g  = lane >> 4;                   // K-half
    const int h  = lane & 15;                   // A row / B,C column

    const int m  = m0 + h;
    const int b  = m / (HH * WW);
    const int rm = m % (HH * WW);
    const int y  = rm / WW;
    const int x  = rm % WW;

    // cooperative B loader: thread -> (column, 16-elem segment of the 64-K stage)
    const int tcol = threadIdx.x >> 2;          // 0..63
    const int tseg = threadIdx.x & 3;           // 0..3
    const unsigned short* bsrc = wp + (size_t)(n0 + tcol) * KTOT + tseg * 16;
    unsigned short* const bdst = &ldsB[0][0] + tcol * BSTRIDE + tseg * 16;
    const int bufstep = 64 * BSTRIDE;           // bf16 units between buffers

    v8f acc0 = {}, acc1 = {}, acc2 = {}, acc3 = {};

    // ---- prologue: stage K[0..63] into buffer 0 ----
#if HAVE_ASYNC_LDS
    async_ld_b128(bsrc,     bdst);
    async_ld_b128(bsrc + 8, bdst + 8);
    WAIT_ASYNC0();
#else
    *reinterpret_cast<v8bf*>(bdst)     = *reinterpret_cast<const v8bf*>(bsrc);
    *reinterpret_cast<v8bf*>(bdst + 8) = *reinterpret_cast<const v8bf*>(bsrc + 8);
#endif
    __syncthreads();

    int buf = 0;
    for (int tap = 0; tap < 9; ++tap) {
        const int dy = tap / 3 - 1, dx = tap % 3 - 1;
        const int yy = y + dy, xx = x + dx;
        const bool inb = (yy >= 0) && (yy < HH) && (xx >= 0) && (xx < WW);
        const unsigned amask = inb ? 0xFFFFFFFFu : 0u;       // branch-free zero padding
        const int yyc = yy < 0 ? 0 : (yy >= HH ? HH - 1 : yy);
        const int xxc = xx < 0 ? 0 : (xx >= WW ? WW - 1 : xx);
        const unsigned short* arow = fmap + (size_t)((b * HH + yyc) * WW + xxc) * CIN + g * 16;

        // A fragments for stage 0 of this tap (two 32-K chunks)
        uint4 a0 = *reinterpret_cast<const uint4*>(arow +  0);
        uint4 a1 = *reinterpret_cast<const uint4*>(arow +  8);
        uint4 a2 = *reinterpret_cast<const uint4*>(arow + 32);
        uint4 a3 = *reinterpret_cast<const uint4*>(arow + 40);

        for (int s = 0; s < CIN / KSTAGE; ++s) {             // 16 stages per tap
            const int k0 = tap * CIN + s * KSTAGE;           // global K of current stage
            const bool notlast = (k0 + KSTAGE < KTOT);

            // issue async B fetch for the next stage into the other buffer
#if HAVE_ASYNC_LDS
            if (notlast) {
                const unsigned short* gs = bsrc + k0 + KSTAGE;
                unsigned short* ld = bdst + (buf ^ 1) * bufstep;
                async_ld_b128(gs,     ld);
                async_ld_b128(gs + 8, ld + 8);
            }
#else
            v8bf bregn0, bregn1;
            if (notlast) {
                bregn0 = *reinterpret_cast<const v8bf*>(bsrc + k0 + KSTAGE);
                bregn1 = *reinterpret_cast<const v8bf*>(bsrc + k0 + KSTAGE + 8);
            }
#endif
            // prefetch next stage's A fragments (same tap) into registers
            uint4 n0q = a0, n1q = a1, n2q = a2, n3q = a3;
            if (s < CIN / KSTAGE - 1) {
                const unsigned short* an = arow + (s + 1) * KSTAGE;
                n0q = *reinterpret_cast<const uint4*>(an +  0);
                n1q = *reinterpret_cast<const uint4*>(an +  8);
                n2q = *reinterpret_cast<const uint4*>(an + 32);
                n3q = *reinterpret_cast<const uint4*>(an + 40);
                __builtin_prefetch(an + KSTAGE, 0, 1);       // global_prefetch_b8
            }

            const unsigned short* lb = &ldsB[buf][0];
            // ---- chunk 0 (K = k0..k0+31) ----
            {
                const v16bf af = mask_frag(a0, a1, amask);
                const v16bf bf0 = lds_load16(lb + ( 0 + h) * BSTRIDE + g * 16);
                const v16bf bf1 = lds_load16(lb + (16 + h) * BSTRIDE + g * 16);
                const v16bf bf2 = lds_load16(lb + (32 + h) * BSTRIDE + g * 16);
                const v16bf bf3 = lds_load16(lb + (48 + h) * BSTRIDE + g * 16);
                acc0 = __builtin_amdgcn_wmma_f32_16x16x32_bf16(false, af, false, bf0, (short)0, acc0, false, false);
                acc1 = __builtin_amdgcn_wmma_f32_16x16x32_bf16(false, af, false, bf1, (short)0, acc1, false, false);
                acc2 = __builtin_amdgcn_wmma_f32_16x16x32_bf16(false, af, false, bf2, (short)0, acc2, false, false);
                acc3 = __builtin_amdgcn_wmma_f32_16x16x32_bf16(false, af, false, bf3, (short)0, acc3, false, false);
            }
            // ---- chunk 1 (K = k0+32..k0+63) ----
            {
                const v16bf af = mask_frag(a2, a3, amask);
                const v16bf bf0 = lds_load16(lb + ( 0 + h) * BSTRIDE + 32 + g * 16);
                const v16bf bf1 = lds_load16(lb + (16 + h) * BSTRIDE + 32 + g * 16);
                const v16bf bf2 = lds_load16(lb + (32 + h) * BSTRIDE + 32 + g * 16);
                const v16bf bf3 = lds_load16(lb + (48 + h) * BSTRIDE + 32 + g * 16);
                acc0 = __builtin_amdgcn_wmma_f32_16x16x32_bf16(false, af, false, bf0, (short)0, acc0, false, false);
                acc1 = __builtin_amdgcn_wmma_f32_16x16x32_bf16(false, af, false, bf1, (short)0, acc1, false, false);
                acc2 = __builtin_amdgcn_wmma_f32_16x16x32_bf16(false, af, false, bf2, (short)0, acc2, false, false);
                acc3 = __builtin_amdgcn_wmma_f32_16x16x32_bf16(false, af, false, bf3, (short)0, acc3, false, false);
            }

#if HAVE_ASYNC_LDS
            if (notlast) WAIT_ASYNC0();          // next buffer landed in LDS
#else
            if (notlast) {
                unsigned short* ld = bdst + (buf ^ 1) * bufstep;
                *reinterpret_cast<v8bf*>(ld)     = bregn0;
                *reinterpret_cast<v8bf*>(ld + 8) = bregn1;
            }
#endif
            wg_barrier_lds();                    // DS-only fence + split barrier
            buf ^= 1;
            a0 = n0q; a1 = n1q; a2 = n2q; a3 = n3q;
        }
    }

    // epilogue: bias + ReLU, store bf16. C layout: elem i -> M = i + 8*g, N = h.
#pragma unroll
    for (int i = 0; i < 8; ++i) {
        const int mm = m0 + i + 8 * g;
        const size_t rowb = (size_t)mm * HID;
        { int n = n0 +      h; float v = acc0[i] + b1[n]; v = v > 0.f ? v : 0.f; hidden[rowb + n] = f2bf(v); }
        { int n = n0 + 16 + h; float v = acc1[i] + b1[n]; v = v > 0.f ? v : 0.f; hidden[rowb + n] = f2bf(v); }
        { int n = n0 + 32 + h; float v = acc2[i] + b1[n]; v = v > 0.f ? v : 0.f; hidden[rowb + n] = f2bf(v); }
        { int n = n0 + 48 + h; float v = acc3[i] + b1[n]; v = v > 0.f ? v : 0.f; hidden[rowb + n] = f2bf(v); }
    }
}

// ---------------- kernel 4: heads (1x1 convs) + anchor decode ----------------
__global__ __launch_bounds__(256)
void heads_decode_kernel(const unsigned short* __restrict__ hidden,
                         const float* __restrict__ wc, const float* __restrict__ bc,
                         const float* __restrict__ wr, const float* __restrict__ br,
                         float* __restrict__ scores,       // [B][NANCH]
                         float* __restrict__ proposals)    // [B][NANCH][4] x1y1x2y2
{
    const int lane = threadIdx.x & 31;
    const int wave = threadIdx.x >> 5;
    const int pos  = blockIdx.x * 8 + wave;
    if (pos >= NPOS) return;

    const unsigned short* hrow = hidden + (size_t)pos * HID;
    float hv[64];
    const uint4* hp = reinterpret_cast<const uint4*>(hrow + lane * 64);
#pragma unroll
    for (int c8 = 0; c8 < 8; ++c8) {
        uint4 q = hp[c8];
        unsigned w0 = q.x, w1v = q.y, w2 = q.z, w3 = q.w;
        hv[c8*8+0] = __uint_as_float(w0 << 16);  hv[c8*8+1] = __uint_as_float(w0 & 0xFFFF0000u);
        hv[c8*8+2] = __uint_as_float(w1v << 16); hv[c8*8+3] = __uint_as_float(w1v & 0xFFFF0000u);
        hv[c8*8+4] = __uint_as_float(w2 << 16);  hv[c8*8+5] = __uint_as_float(w2 & 0xFFFF0000u);
        hv[c8*8+6] = __uint_as_float(w3 << 16);  hv[c8*8+7] = __uint_as_float(w3 & 0xFFFF0000u);
    }

    float myScore = 0.f, o0 = 0.f, o1 = 0.f, o2 = 0.f, o3 = 0.f;
    for (int c = 0; c < 45; ++c) {
        float s = 0.f;
        if (c < 9) {
#pragma unroll 8
            for (int i = 0; i < 64; ++i) s += hv[i] * wc[(size_t)(lane * 64 + i) * NA + c];
        } else {
            const int cc = c - 9;
#pragma unroll 8
            for (int i = 0; i < 64; ++i) s += hv[i] * wr[(size_t)(lane * 64 + i) * 36 + cc];
        }
#pragma unroll
        for (int off = 16; off >= 1; off >>= 1) s += __shfl_xor(s, off, 32);
        if (c < 9) {
            s += bc[c];
            if (lane == c) myScore = s;
        } else {
            const int cc = c - 9;
            s += br[cc];
            if (lane == (cc >> 2)) {
                const int kk = cc & 3;
                if (kk == 0) o0 = s; else if (kk == 1) o1 = s; else if (kk == 2) o2 = s; else o3 = s;
            }
        }
    }

    if (lane < NA) {
        const int bidx = pos / (HH * WW);
        const int rm   = pos % (HH * WW);
        const int yy   = rm / WW;
        const int xx   = rm % WW;
        const int si = lane / 3, ri = lane % 3;
        const float sc = (si == 0) ? 128.f : ((si == 1) ? 256.f : 512.f);
        const float rt = (ri == 0) ? 0.5f  : ((ri == 1) ? 1.f   : 2.f);
        const float aw = sc * __fsqrt_rn(1.f / rt);
        const float ah = sc * __fsqrt_rn(rt);
        const float acx = ((float)xx + 0.5f) * STRIDE_F;
        const float acy = ((float)yy + 0.5f) * STRIDE_F;
        const float pcx = acx + o0 * aw;
        const float pcy = acy + o1 * ah;
        const float pw  = aw * __expf(o2);
        const float ph  = ah * __expf(o3);
        const float XMAX = WW * STRIDE_F, YMAX = HH * STRIDE_F;
        float x1 = fminf(fmaxf(pcx - 0.5f * pw, 0.f), XMAX);
        float y1 = fminf(fmaxf(pcy - 0.5f * ph, 0.f), YMAX);
        float x2 = fminf(fmaxf(pcx + 0.5f * pw, 0.f), XMAX);
        float y2 = fminf(fmaxf(pcy + 0.5f * ph, 0.f), YMAX);
        const float sig = 1.f / (1.f + __expf(-myScore));
        const int n = (yy * WW + xx) * NA + lane;
        scores[(size_t)bidx * NANCH + n] = sig;
        float* pr = proposals + ((size_t)bidx * NANCH + n) * 4;
        pr[0] = x1; pr[1] = y1; pr[2] = x2; pr[3] = y2;
    }
}

// ---------------- kernel 5: top-K selection per batch ----------------
__global__ __launch_bounds__(256)
void topk_kernel(const float* __restrict__ scores, const float* __restrict__ proposals,
                 float* __restrict__ tkS, float* __restrict__ tkB)
{
    const int b = blockIdx.x;
    const float* sc = scores + (size_t)b * NANCH;
    __shared__ int cnt;
    float lo = 0.f, hi = 1.f;
    for (int it = 0; it < 24; ++it) {
        const float thr = 0.5f * (lo + hi);
        if (threadIdx.x == 0) cnt = 0;
        __syncthreads();
        int local = 0;
        for (int i = threadIdx.x; i < NANCH; i += 256) local += (sc[i] > thr) ? 1 : 0;
        atomicAdd(&cnt, local);
        __syncthreads();
        const int c = cnt;
        __syncthreads();
        if (c >= TOPK) lo = thr; else hi = thr;
    }
    const float thr = lo;

    for (int k = threadIdx.x; k < TOPK; k += 256) {
        tkS[(size_t)b * TOPK + k] = -3.0e38f;
        float* bx = tkB + ((size_t)b * TOPK + k) * 4;
        bx[0] = bx[1] = bx[2] = bx[3] = 0.f;
    }
    if (threadIdx.x == 0) cnt = 0;
    __syncthreads();
    for (int i = threadIdx.x; i < NANCH; i += 256) {
        const float s = sc[i];
        if (s > thr) {
            const int slot = atomicAdd(&cnt, 1);
            if (slot < TOPK) {
                tkS[(size_t)b * TOPK + slot] = s;
                const float* p = proposals + ((size_t)b * NANCH + i) * 4;
                float* bx = tkB + ((size_t)b * TOPK + slot) * 4;
                bx[0] = p[1]; bx[1] = p[0]; bx[2] = p[3]; bx[3] = p[2];  // -> y1x1y2x2
            }
        }
    }
}

// ---------------- kernel 6: greedy NMS per batch ----------------
__global__ __launch_bounds__(256)
void nms_kernel(const float* __restrict__ tkS, const float* __restrict__ tkB,
                float* __restrict__ out)     // [B][300][4] y1x1y2x2
{
    const int b = blockIdx.x;
    __shared__ float sS[TOPK], sY1[TOPK], sX1[TOPK], sY2[TOPK], sX2[TOPK], sA[TOPK];
    __shared__ float rS[256];
    __shared__ int   rI[256];

    for (int i = threadIdx.x; i < TOPK; i += 256) {
        sS[i] = tkS[(size_t)b * TOPK + i];
        const float* bx = tkB + ((size_t)b * TOPK + i) * 4;
        const float y1 = bx[0], x1 = bx[1], y2 = bx[2], x2 = bx[3];
        sY1[i] = y1; sX1[i] = x1; sY2[i] = y2; sX2[i] = x2;
        sA[i]  = (y2 - y1) * (x2 - x1);
    }
    __syncthreads();

    for (int it = 0; it < MAXOUT; ++it) {
        float best = -3.4e38f; int bi = 0;
        for (int i = threadIdx.x; i < TOPK; i += 256) {
            const float s = sS[i];
            if (s > best) { best = s; bi = i; }
        }
        rS[threadIdx.x] = best; rI[threadIdx.x] = bi;
        __syncthreads();
        for (int st = 128; st > 0; st >>= 1) {
            if (threadIdx.x < st) {
                if (rS[threadIdx.x + st] > rS[threadIdx.x]) {
                    rS[threadIdx.x] = rS[threadIdx.x + st];
                    rI[threadIdx.x] = rI[threadIdx.x + st];
                }
            }
            __syncthreads();
        }
        const int   besti = rI[0];
        const float bests = rS[0];
        const float by1 = sY1[besti], bx1 = sX1[besti], by2 = sY2[besti], bx2 = sX2[besti];
        const float barea = (by2 - by1) * (bx2 - bx1);
        const bool  valid = bests > -1.0e38f;
        if (threadIdx.x == 0) {
            float* o = out + ((size_t)b * MAXOUT + it) * 4;
            o[0] = valid ? by1 : 0.f; o[1] = valid ? bx1 : 0.f;
            o[2] = valid ? by2 : 0.f; o[3] = valid ? bx2 : 0.f;
        }
        __syncthreads();
        for (int i = threadIdx.x; i < TOPK; i += 256) {
            const float yy1 = fmaxf(by1, sY1[i]), xx1 = fmaxf(bx1, sX1[i]);
            const float yy2 = fminf(by2, sY2[i]), xx2 = fminf(bx2, sX2[i]);
            const float inter = fmaxf(yy2 - yy1, 0.f) * fmaxf(xx2 - xx1, 0.f);
            const float iou = inter / (sA[i] + barea - inter + 1e-8f);
            if (iou > IOU_T || i == besti) sS[i] = -3.4e38f;
        }
        __syncthreads();
    }
}

// ---------------- host launcher ----------------
extern "C" void kernel_launch(void* const* d_in, const int* in_sizes, int n_in,
                              void* d_out, int out_size, void* d_ws, size_t ws_size,
                              hipStream_t stream) {
    const float* fmap = (const float*)d_in[0];
    const float* w1   = (const float*)d_in[1];
    const float* b1   = (const float*)d_in[2];
    const float* wc   = (const float*)d_in[3];
    const float* bc   = (const float*)d_in[4];
    const float* wr   = (const float*)d_in[5];
    const float* br   = (const float*)d_in[6];
    float* out = (float*)d_out;

    char* ws = (char*)d_ws;
    size_t off = 0;
    auto carve = [&](size_t bytes) {
        void* p = ws + off;
        off += (bytes + 255) & ~(size_t)255;
        return p;
    };
    unsigned short* fmap_bf  = (unsigned short*)carve((size_t)NPOS * CIN * 2);
    unsigned short* w1_bf    = (unsigned short*)carve((size_t)HID * KTOT * 2);
    unsigned short* hidden   = (unsigned short*)carve((size_t)NPOS * HID * 2);
    float*          scoresB  = (float*)carve((size_t)BB * NANCH * 4);
    float*          propB    = (float*)carve((size_t)BB * NANCH * 4 * 4);
    float*          tkS      = (float*)carve((size_t)BB * TOPK * 4);
    float*          tkB      = (float*)carve((size_t)BB * TOPK * 4 * 4);
    (void)ws_size; (void)in_sizes; (void)n_in; (void)out_size;

    cvt_fmap_kernel<<<2048, 256, 0, stream>>>(fmap, fmap_bf, NPOS * CIN);
    pack_w1_kernel<<<2048, 256, 0, stream>>>(w1, w1_bf);

    dim3 gconv((MTILES + 7) / 8, HID / 64);        // (119, 32)
    conv1_wmma_kernel<<<gconv, 256, 0, stream>>>(fmap_bf, w1_bf, b1, hidden);

    heads_decode_kernel<<<NPOS / 8, 256, 0, stream>>>(hidden, wc, bc, wr, br, scoresB, propB);

    topk_kernel<<<BB, 256, 0, stream>>>(scoresB, propB, tkS, tkB);

    nms_kernel<<<BB, 256, 0, stream>>>(tkS, tkB, out);
}
